// LocalWindowAttn_47261820125932
// MI455X (gfx1250) — compile-verified, matrix operand order checked by reading, not run on hardware
//
#include <hip/hip_runtime.h>

typedef __attribute__((ext_vector_type(16))) __bf16 v16bf;
typedef __attribute__((ext_vector_type(8)))  __bf16 v8bf;
typedef __attribute__((ext_vector_type(4)))  __bf16 v4bf;
typedef __attribute__((ext_vector_type(8)))  float  v8f;

#define WMMA_BF16(a, b, c) \
  __builtin_amdgcn_wmma_f32_16x16x32_bf16(false, (a), false, (b), (short)0, (c), false, false)

// ---------------------------------------------------------------------------
// 16-bit fragment loader (wave32, CDNA5 ISA 7.12.2 layout):
//   lanes 0-15 : M/N = lane,    K = {0..7} U {16..23}
//   lanes 16-31: M/N = lane-16, K = {8..15} U {24..31}
// Works as: A fragment of row-major [M][K]; B fragment where B(k,n)=mat[n][k].
// Generic pointer: compiler infers LDS (ds_load_b128) or global
// (global_load_b128) after inlining.
// ---------------------------------------------------------------------------
__device__ __forceinline__ v16bf load_mat16(const __bf16* base, int ld, int lane) {
  const int m   = lane & 15;
  const int off = (lane & 16) ? 8 : 0;
  const __bf16* row = base + (size_t)m * ld + off;
  v8bf lo = *(const v8bf*)(row);        // K = off .. off+7
  v8bf hi = *(const v8bf*)(row + 16);   // K = 16+off .. 16+off+7
  return __builtin_shufflevector(lo, hi, 0, 1, 2, 3, 4, 5, 6, 7,
                                         8, 9, 10, 11, 12, 13, 14, 15);
}

// ---------------------------------------------------------------------------
// Prep: convert fp32 weights -> bf16 into workspace (once per launch, so the
// 2048 attention workgroups never touch fp32 weights or pay cvt VALU cost).
//   d_ws layout: [0 .. 196607]  in_proj_w (768x256)
//                [196608 .. ]   out_w     (256x256)
// ---------------------------------------------------------------------------
__global__ __launch_bounds__(256)
void cvt_weights_kernel(const float* __restrict__ w_in,
                        const float* __restrict__ w_out,
                        __bf16* __restrict__ dst) {
  const int i = (blockIdx.x * 256 + threadIdx.x) * 4;  // 262144 elems total
  const float* src = (i < 196608) ? (w_in + i) : (w_out + (i - 196608));
  const float4 f = *(const float4*)src;
  v4bf o;
  o[0] = (__bf16)f.x; o[1] = (__bf16)f.y; o[2] = (__bf16)f.z; o[3] = (__bf16)f.w;
  *(v4bf*)(dst + i) = o;
}

// ---------------------------------------------------------------------------
// One workgroup per 7x7 window. 256 threads = 8 waves.
// ---------------------------------------------------------------------------
__global__ __launch_bounds__(256, 1)
void swin_window_attn_kernel(const float* __restrict__ x,
                             const __bf16* __restrict__ w_in_bf,   // [768][256]
                             const float* __restrict__ b_in,
                             const __bf16* __restrict__ w_out_bf,  // [256][256]
                             const float* __restrict__ b_out,
                             float* __restrict__ out) {
  constexpr int D = 256, HW = 56, WSZ = 7, LP = 64, LV = 49, HD = 32;
  constexpr int VTLD = 72;  // padded ld for V^T: 16B lane reads hit disjoint banks

  // 148 KB LDS: X (reused as O) | Q | K | V^T | P scratch
  __shared__ __align__(16) __bf16 smem[3 * LP * D + D * VTLD + 8 * 16 * LP];
  __bf16* sh_x  = smem;                          // [64][256] bf16 (X, later O)
  __bf16* sh_q  = smem + 1 * LP * D;             // [64][256] pre-scaled + bias
  __bf16* sh_k  = smem + 2 * LP * D;             // [64][256]
  __bf16* sh_vt = smem + 3 * LP * D;             // [256][72]  V transposed
  __bf16* sh_p  = smem + 3 * LP * D + D * VTLD;  // [8][16][64]

  const int tid    = threadIdx.x;
  const int lane   = tid & 31;
  const int wave   = tid >> 5;
  const int lane15 = lane & 15;
  const int laneHi = (lane >> 4) & 1;

  const int blk  = blockIdx.x;  // 2048 = 32 * 8 * 8
  const int bi   = blk >> 6;
  const int wh   = (blk >> 3) & 7;
  const int wwc  = blk & 7;
  const int row0 = wh * WSZ, col0 = wwc * WSZ;

  // ---- Phase 1: window X -> LDS bf16, zero-pad rows 49..63 ----------------
  for (int idx = tid; idx < LP * (D / 4); idx += 256) {
    const int l  = idx >> 6;
    const int d4 = (idx & 63) * 4;
    __bf16* dst = sh_x + l * D + d4;
    if (l < LV) {
      const int gr = row0 + l / WSZ, gc = col0 + l % WSZ;
      const float4 f = *(const float4*)(x + (((size_t)bi * HW + gr) * HW + gc) * D + d4);
      v4bf o;
      o[0] = (__bf16)f.x; o[1] = (__bf16)f.y; o[2] = (__bf16)f.z; o[3] = (__bf16)f.w;
      *(v4bf*)dst = o;
    } else {
      v4bf z = {};
      *(v4bf*)dst = z;
    }
  }
  __syncthreads();

  // ---- Phase 2: QKV = X @ Wqkv^T + b ------------------------------------
  // Each wave owns e-tiles; B fragment loaded once per K-step, reused over
  // all 4 row tiles (4 accumulators).
  const float SCALE = 0.17677669529663687f;  // 1/sqrt(32)
  for (int et = wave; et < 48; et += 8) {
    const int e0 = et << 4;  // 0..752
    v8f acc[4] = {{}, {}, {}, {}};
#pragma unroll
    for (int kk = 0; kk < 8; ++kk) {
      v16bf b = load_mat16(w_in_bf + (size_t)e0 * D + kk * 32, D, lane);
#pragma unroll
      for (int i = 0; i < 4; ++i) {
        v16bf a = load_mat16(sh_x + (i * 16) * D + kk * 32, D, lane);
        acc[i] = WMMA_BF16(a, b, acc[i]);
      }
    }
    const float bias = b_in[e0 + lane15];
    if (e0 < 512) {                      // Q or K: row-major [l][256]
      const bool isQ = (e0 < 256);
      __bf16* dst = isQ ? sh_q : sh_k;
      const int col = (e0 & 255) + lane15;
#pragma unroll
      for (int i = 0; i < 4; ++i) {
        const int mb = i * 16 + 8 * laneHi;
#pragma unroll
        for (int r = 0; r < 8; ++r) {
          float v = acc[i][r] + bias;
          if (isQ) v *= SCALE;
          dst[(mb + r) * D + col] = (__bf16)v;
        }
      }
    } else {                             // V: store transposed [d][72]
      const int col = (e0 - 512) + lane15;
#pragma unroll
      for (int i = 0; i < 4; ++i) {
        const int mb = i * 16 + 8 * laneHi;
        v8bf vv;
#pragma unroll
        for (int r = 0; r < 8; ++r) vv[r] = (__bf16)(acc[i][r] + bias);
        *(v8bf*)(sh_vt + (size_t)col * VTLD + mb) = vv;   // one b128 store
      }
    }
  }
  __syncthreads();

  // ---- Phase 3: per-head attention (wave == head) -------------------------
  {
    const int h = wave;
    __bf16* pb = sh_p + h * 16 * LP;
    for (int i = 0; i < 4; ++i) {
      // S = (scaled Q) @ K^T : one WMMA per 16x16 tile (K = hd = 32)
      v16bf aq = load_mat16(sh_q + (i * 16) * D + h * HD, D, lane);
      v8f s[4];
#pragma unroll
      for (int j = 0; j < 4; ++j) {
        v16bf bk = load_mat16(sh_k + (j * 16) * D + h * HD, D, lane);  // B^T == A layout
        v8f z = {};
        s[j] = WMMA_BF16(aq, bk, z);
      }
      // mask key columns >= 49
#pragma unroll
      for (int j = 0; j < 4; ++j) {
        if (j * 16 + lane15 >= LV) {
#pragma unroll
          for (int r = 0; r < 8; ++r) s[j][r] = -1.0e30f;
        }
      }
      // row softmax: per-lane combine over 4 col fragments, then 16-lane xor
#pragma unroll
      for (int r = 0; r < 8; ++r) {
        float mx = fmaxf(fmaxf(s[0][r], s[1][r]), fmaxf(s[2][r], s[3][r]));
#pragma unroll
        for (int dm = 1; dm < 16; dm <<= 1) mx = fmaxf(mx, __shfl_xor(mx, dm, 32));
        const float p0 = __expf(s[0][r] - mx), p1 = __expf(s[1][r] - mx);
        const float p2 = __expf(s[2][r] - mx), p3 = __expf(s[3][r] - mx);
        float sum = (p0 + p1) + (p2 + p3);
#pragma unroll
        for (int dm = 1; dm < 16; dm <<= 1) sum += __shfl_xor(sum, dm, 32);
        const float rs = 1.0f / sum;
        const int m = r + 8 * laneHi;
        pb[m * LP +  0 + lane15] = (__bf16)(p0 * rs);
        pb[m * LP + 16 + lane15] = (__bf16)(p1 * rs);
        pb[m * LP + 32 + lane15] = (__bf16)(p2 * rs);
        pb[m * LP + 48 + lane15] = (__bf16)(p3 * rs);
      }
      asm volatile("s_wait_dscnt 0" ::: "memory");  // same-wave LDS RAW fence

      // O tile = P @ V : V^T gives contiguous-K B fragments
#pragma unroll
      for (int c = 0; c < 2; ++c) {
        v8f acc = {};
#pragma unroll
        for (int kk = 0; kk < 2; ++kk) {
          v16bf ap = load_mat16(pb + kk * 32, LP, lane);
          v16bf bv = load_mat16(sh_vt + (size_t)(h * HD + c * 16) * VTLD + kk * 32,
                                VTLD, lane);
          acc = WMMA_BF16(ap, bv, acc);
        }
        const int mb = i * 16 + 8 * laneHi;
#pragma unroll
        for (int r = 0; r < 8; ++r)
          sh_x[(mb + r) * D + h * HD + c * 16 + lane15] = (__bf16)acc[r];  // O reuses X
      }
      asm volatile("s_wait_dscnt 0" ::: "memory");
    }
  }
  __syncthreads();

  // ---- Phase 4: Y = O @ Wout^T + b, store rows < 49 -----------------------
  for (int et = wave; et < 16; et += 8) {
    const int e0 = et << 4;
    v8f acc[4] = {{}, {}, {}, {}};
#pragma unroll
    for (int kk = 0; kk < 8; ++kk) {
      v16bf b = load_mat16(w_out_bf + (size_t)e0 * D + kk * 32, D, lane);
#pragma unroll
      for (int i = 0; i < 4; ++i) {
        v16bf a = load_mat16(sh_x + (i * 16) * D + kk * 32, D, lane);
        acc[i] = WMMA_BF16(a, b, acc[i]);
      }
    }
    const float bias = b_out[e0 + lane15];
#pragma unroll
    for (int i = 0; i < 4; ++i) {
      const int mb = i * 16 + 8 * laneHi;
#pragma unroll
      for (int r = 0; r < 8; ++r) {
        const int l = mb + r;
        if (l < LV) {
          const int gr = row0 + l / WSZ, gc = col0 + l % WSZ;
          out[(((size_t)bi * HW + gr) * HW + gc) * D + e0 + lane15] = acc[i][r] + bias;
        }
      }
    }
  }
}

extern "C" void kernel_launch(void* const* d_in, const int* in_sizes, int n_in,
                              void* d_out, int out_size, void* d_ws, size_t ws_size,
                              hipStream_t stream) {
  (void)in_sizes; (void)n_in; (void)out_size; (void)ws_size;
  const float* x     = (const float*)d_in[0];
  const float* w_in  = (const float*)d_in[1];
  const float* b_in  = (const float*)d_in[2];
  const float* w_out = (const float*)d_in[3];
  const float* b_out = (const float*)d_in[4];
  float* out = (float*)d_out;

  __bf16* w_bf = (__bf16*)d_ws;  // 262144 bf16 = 512 KB scratch
  // 262144 elems / 4 per thread / 256 threads = 256 blocks
  cvt_weights_kernel<<<256, 256, 0, stream>>>(w_in, w_out, w_bf);

  dim3 grid(2048);   // 32 batches * 8 * 8 windows
  dim3 block(256);   // 8 wave32s
  swin_window_attn_kernel<<<grid, block, 0, stream>>>(
      x, w_bf, b_in, w_bf + 196608, b_out, out);
}
